// Net_16484084483040
// MI455X (gfx1250) — matrix operand order repaired and checked by reading, not compiled, not run on hardware
//
#include <hip/hip_runtime.h>
#include <hip/hip_bf16.h>

typedef __attribute__((ext_vector_type(16))) __bf16 v16bf;
typedef __attribute__((ext_vector_type(8)))  __bf16 v8bf;
typedef __attribute__((ext_vector_type(8)))  float  v8f;

union BF16Frag { v16bf v; v8bf h[2]; };

// A-matrix 16x32 bf16 fragment: per-lane 16 values are k = kb+half*8 .. +7 and
// kb+16+half*8 .. +7 of row M (row-major source, 16B-aligned chunks).
__device__ __forceinline__ v16bf load_a_frag(const __bf16* base) {
    BF16Frag f;
    f.h[0] = *(const v8bf*)(base);
    f.h[1] = *(const v8bf*)(base + 16);
    return f.v;
}
// B-matrix 32x16 bf16 fragment: per-lane k = kb+half*8 .. +7 and kb+16+half*8 .. +7
// of column N (from a K-major transposed weight buffer).
__device__ __forceinline__ v16bf load_b_frag(const __bf16* base) {
    BF16Frag f;
    f.h[0] = *(const v8bf*)(base);
    f.h[1] = *(const v8bf*)(base + 16);
    return f.v;
}

__device__ __forceinline__ v8f wmma_bf16(v16bf a, v16bf b, v8f c) {
    return __builtin_amdgcn_wmma_f32_16x16x32_bf16(
        false, a, false, b, (short)0, c, false, false);
}

__global__ __launch_bounds__(256) void zero_f32(float* __restrict__ p, long long n) {
    long long i = (long long)blockIdx.x * blockDim.x + threadIdx.x;
    long long stride = (long long)gridDim.x * blockDim.x;
    for (; i < n; i += stride) p[i] = 0.0f;
}

// Fold BN coefficients, convert + transpose weights to bf16 K-major.
__global__ __launch_bounds__(256) void prep_kernel(
    const float* __restrict__ w1, const float* __restrict__ b1,
    const float* __restrict__ g1, const float* __restrict__ be1,
    const float* __restrict__ w2, const float* __restrict__ w3,
    const float* __restrict__ b3, const float* __restrict__ g2,
    const float* __restrict__ be2, const float* __restrict__ w4,
    const float* __restrict__ wl1,
    float* __restrict__ a1, float* __restrict__ b1f,
    float* __restrict__ s3, float* __restrict__ c3,
    __bf16* __restrict__ w2T, __bf16* __restrict__ w3T,
    __bf16* __restrict__ w4T, __bf16* __restrict__ wl1T)
{
    const float inv = rsqrtf(1.0f + 1e-5f);
    int gid = blockIdx.x * blockDim.x + threadIdx.x;
    int stride = gridDim.x * blockDim.x;
    for (int j = gid; j < 512; j += stride) {
        float s = g1[j] * inv;
        a1[j]  = s * w1[j];             // w1 is [1,512]
        b1f[j] = s * b1[j] + be1[j];
    }
    for (int j = gid; j < 128; j += stride) {
        float s = g2[j] * inv;
        s3[j] = s;
        c3[j] = s * b3[j] + be2[j];
    }
    for (int i = gid; i < 512 * 256; i += stride) {   // w2T[256][512]
        int n = i >> 9, k = i & 511;
        w2T[i] = (__bf16)w2[(size_t)k * 256 + n];
    }
    for (int i = gid; i < 256 * 128; i += stride) {   // w3T[128][256]
        int n = i >> 8, k = i & 255;
        w3T[i] = (__bf16)w3[(size_t)k * 128 + n];
    }
    for (int i = gid; i < 128 * 64; i += stride) {    // w4T[64][128]
        int n = i >> 7, k = i & 127;
        w4T[i] = (__bf16)w4[(size_t)k * 64 + n];
    }
    for (int i = gid; i < 64 * 16; i += stride) {     // wl1T[16][64]
        int n = i >> 6, k = i & 63;
        wl1T[i] = (__bf16)wl1[(size_t)k * 16 + n];
    }
}

// agg1: s1[dst] += x[src]   (scalar feature)
__global__ __launch_bounds__(256) void agg1_kernel(
    const float* __restrict__ x, const long long* __restrict__ ei,
    float* __restrict__ s1, int E)
{
    int e = blockIdx.x * blockDim.x + threadIdx.x;
    if (e < E) {
        int s = (int)ei[e];
        int d = (int)ei[(size_t)E + e];
        atomicAdd(&s1[d], x[s]);
    }
}

// Layer1 fused + GEMM 512->256 (bf16 WMMA). 32-node macro-tile per workgroup:
// each B fragment load feeds two WMMAs (two node row-tiles).
__global__ __launch_bounds__(256) void mlp1_gemm_kernel(
    const float* __restrict__ x, const float* __restrict__ s1,
    const float* __restrict__ a1, const float* __restrict__ b1f,
    const float* __restrict__ b2, const __bf16* __restrict__ w2T,
    __bf16* __restrict__ h2, int N)
{
    __shared__ __bf16 row1[32 * 512];   // 32KB: relu(BN(Linear1)) activations
    __shared__ float  tval[32];
    const int tid = threadIdx.x;
    const int node0 = blockIdx.x * 32;

    if (tid < 32) tval[tid] = x[node0 + tid] + s1[node0 + tid];
    __syncthreads();

    for (int idx = tid; idx < 32 * 512; idx += 256) {
        int m = idx >> 9, j = idx & 511;
        float v = fmaf(tval[m], a1[j], b1f[j]);
        row1[idx] = (__bf16)fmaxf(v, 0.0f);
    }
    __syncthreads();

    const int wave = tid >> 5, lane = tid & 31;
    const int M = lane & 15, half = lane >> 4;

    for (int t = 0; t < 2; ++t) {
        const int col = (wave * 2 + t) * 16 + (lane & 15);
        const __bf16* brow = w2T + (size_t)col * 512;
        v8f acc0 = {}, acc1 = {};
        #pragma unroll 4
        for (int kb = 0; kb < 512; kb += 32) {
            v16bf b  = load_b_frag(brow + kb + half * 8);
            v16bf a0 = load_a_frag(&row1[M * 512 + kb + half * 8]);
            v16bf a1f = load_a_frag(&row1[(16 + M) * 512 + kb + half * 8]);
            acc0 = wmma_bf16(a0, b, acc0);
            acc1 = wmma_bf16(a1f, b, acc1);
        }
        const float bias = b2[col];
        #pragma unroll
        for (int r = 0; r < 8; ++r) {
            int row = node0 + r + half * 8;
            h2[(size_t)row * 256 + col] = (__bf16)(acc0[r] + bias);
            h2[(size_t)(row + 16) * 256 + col] = (__bf16)(acc1[r] + bias);
        }
    }
}

// agg2: s2[dst][:] += h2[src][:]  — one wave per edge, 8 channels/lane.
__global__ __launch_bounds__(256) void agg2_kernel(
    const __bf16* __restrict__ h2, const long long* __restrict__ ei,
    float* __restrict__ s2, int E)
{
    const int wid = (int)((blockIdx.x * blockDim.x + threadIdx.x) >> 5);
    const int lane = threadIdx.x & 31;
    if (wid >= E) return;
    const int s = (int)ei[wid];
    const int d = (int)ei[(size_t)E + wid];
    const __bf16* hs = h2 + (size_t)s * 256;
    float* outp = s2 + (size_t)d * 256;
    #pragma unroll
    for (int c8 = 0; c8 < 8; ++c8) {
        int c = c8 * 32 + lane;
        atomicAdd(&outp[c], (float)hs[c]);
    }
}

// Fused tail: (h2+s2) -> 128 (BN+ReLU) -> 64 -> 16 (ReLU) -> 6 -> log_softmax.
// 32-node macro-tile per workgroup; B fragments reused across two node tiles.
__global__ __launch_bounds__(256) void tail_kernel(
    const __bf16* __restrict__ h2, const float* __restrict__ s2,
    const float* __restrict__ s3, const float* __restrict__ c3,
    const float* __restrict__ b4,
    const __bf16* __restrict__ w3T, const __bf16* __restrict__ w4T,
    const __bf16* __restrict__ wl1T,
    const float* __restrict__ bl1, const float* __restrict__ wl2,
    const float* __restrict__ bl2, float* __restrict__ out, int N)
{
    __shared__ __bf16 hin[32 * 256];   // 16KB
    __shared__ __bf16 h3[32 * 128];    // 8KB
    __shared__ __bf16 h4[32 * 64];     // 4KB
    __shared__ float  h5[32 * 16];     // 2KB
    const int tid = threadIdx.x;
    const int node0 = blockIdx.x * 32;

    for (int idx = tid; idx < 32 * 256; idx += 256) {
        int m = idx >> 8, c = idx & 255;
        size_t g = (size_t)(node0 + m) * 256 + c;
        hin[idx] = (__bf16)((float)h2[g] + s2[g]);
    }
    __syncthreads();

    const int wave = tid >> 5, lane = tid & 31;
    const int M = lane & 15, half = lane >> 4;

    // 256 -> 128 with folded BN + ReLU: wave = col tile, both node tiles.
    {
        const int col = wave * 16 + (lane & 15);
        const __bf16* brow = w3T + (size_t)col * 256;
        v8f acc0 = {}, acc1 = {};
        #pragma unroll
        for (int kb = 0; kb < 256; kb += 32) {
            v16bf b  = load_b_frag(brow + kb + half * 8);
            v16bf a0 = load_a_frag(&hin[M * 256 + kb + half * 8]);
            v16bf a1f = load_a_frag(&hin[(16 + M) * 256 + kb + half * 8]);
            acc0 = wmma_bf16(a0, b, acc0);
            acc1 = wmma_bf16(a1f, b, acc1);
        }
        const float sc = s3[col], cc = c3[col];
        #pragma unroll
        for (int r = 0; r < 8; ++r) {
            float v0 = fmaxf(fmaf(sc, acc0[r], cc), 0.0f);
            float v1 = fmaxf(fmaf(sc, acc1[r], cc), 0.0f);
            h3[(r + half * 8) * 128 + col] = (__bf16)v0;
            h3[(16 + r + half * 8) * 128 + col] = (__bf16)v1;
        }
    }
    __syncthreads();

    // 128 -> 64: 4 col tiles x 2 node tiles = 8 jobs, one per wave.
    {
        const int col = (wave & 3) * 16 + (lane & 15);
        const int mrow = (wave >> 2) * 16;
        const __bf16* brow = w4T + (size_t)col * 128;
        v8f acc = {};
        #pragma unroll
        for (int kb = 0; kb < 128; kb += 32) {
            v16bf b = load_b_frag(brow + kb + half * 8);
            v16bf a = load_a_frag(&h3[(mrow + M) * 128 + kb + half * 8]);
            acc = wmma_bf16(a, b, acc);
        }
        const float bias = b4[col];
        #pragma unroll
        for (int r = 0; r < 8; ++r)
            h4[(mrow + r + half * 8) * 64 + col] = (__bf16)(acc[r] + bias);
    }
    __syncthreads();

    // 64 -> 16 + ReLU: 1 col tile x 2 node tiles, waves 0-1.
    if (wave < 2) {
        const int col = lane & 15;
        const int mrow = wave * 16;
        const __bf16* brow = wl1T + (size_t)col * 64;
        v8f acc = {};
        #pragma unroll
        for (int kb = 0; kb < 64; kb += 32) {
            v16bf b = load_b_frag(brow + kb + half * 8);
            v16bf a = load_a_frag(&h4[(mrow + M) * 64 + kb + half * 8]);
            acc = wmma_bf16(a, b, acc);
        }
        const float bias = bl1[col];
        #pragma unroll
        for (int r = 0; r < 8; ++r)
            h5[(mrow + r + half * 8) * 16 + col] = fmaxf(acc[r] + bias, 0.0f);
    }
    __syncthreads();

    // 16 -> 6 + log_softmax (one thread per node)
    if (tid < 32) {
        float z[6];
        #pragma unroll
        for (int c = 0; c < 6; ++c) {
            float a = bl2[c];
            #pragma unroll
            for (int k = 0; k < 16; ++k)
                a = fmaf(h5[tid * 16 + k], wl2[k * 6 + c], a);
            z[c] = a;
        }
        float mx = z[0];
        #pragma unroll
        for (int c = 1; c < 6; ++c) mx = fmaxf(mx, z[c]);
        float se = 0.0f;
        #pragma unroll
        for (int c = 0; c < 6; ++c) se += expf(z[c] - mx);
        const float ls = logf(se);
        #pragma unroll
        for (int c = 0; c < 6; ++c)
            out[(size_t)(node0 + tid) * 6 + c] = z[c] - mx - ls;
    }
}

extern "C" void kernel_launch(void* const* d_in, const int* in_sizes, int n_in,
                              void* d_out, int out_size, void* d_ws, size_t ws_size,
                              hipStream_t stream) {
    const float*      x   = (const float*)d_in[0];
    const long long*  ei  = (const long long*)d_in[1];   // int64 [2,E]
    const float* w1  = (const float*)d_in[2];
    const float* b1  = (const float*)d_in[3];
    const float* g1  = (const float*)d_in[4];
    const float* be1 = (const float*)d_in[5];
    const float* w2  = (const float*)d_in[6];
    const float* b2  = (const float*)d_in[7];
    const float* w3  = (const float*)d_in[8];
    const float* b3  = (const float*)d_in[9];
    const float* g2  = (const float*)d_in[10];
    const float* be2 = (const float*)d_in[11];
    const float* w4  = (const float*)d_in[12];
    const float* b4  = (const float*)d_in[13];
    const float* wl1 = (const float*)d_in[14];
    const float* bl1 = (const float*)d_in[15];
    const float* wl2 = (const float*)d_in[16];
    const float* bl2 = (const float*)d_in[17];
    float* out = (float*)d_out;

    const int N = in_sizes[0];      // 100000
    const int E = in_sizes[1] / 2;  // 1600000

    char* ws = (char*)d_ws;
    size_t off = 0;
    auto take = [&](size_t bytes) {
        size_t o = off;
        off += (bytes + 255) & ~(size_t)255;
        return o;
    };
    float*  s1   = (float*) (ws + take((size_t)N * 4));
    __bf16* h2   = (__bf16*)(ws + take((size_t)N * 256 * 2));
    float*  s2   = (float*) (ws + take((size_t)N * 256 * 4));
    __bf16* w2T  = (__bf16*)(ws + take(512 * 256 * 2));
    __bf16* w3T  = (__bf16*)(ws + take(256 * 128 * 2));
    __bf16* w4T  = (__bf16*)(ws + take(128 * 64 * 2));
    __bf16* wl1T = (__bf16*)(ws + take(64 * 16 * 2));
    float*  a1   = (float*) (ws + take(512 * 4));
    float*  b1f  = (float*) (ws + take(512 * 4));
    float*  s3   = (float*) (ws + take(128 * 4));
    float*  c3   = (float*) (ws + take(128 * 4));
    (void)ws_size; (void)n_in; (void)out_size;

    zero_f32<<<4096, 256, 0, stream>>>(s2, (long long)N * 256);
    zero_f32<<<256, 256, 0, stream>>>(s1, (long long)N);
    prep_kernel<<<512, 256, 0, stream>>>(w1, b1, g1, be1, w2, w3, b3, g2, be2,
                                         w4, wl1, a1, b1f, s3, c3,
                                         w2T, w3T, w4T, wl1T);
    agg1_kernel<<<(E + 255) / 256, 256, 0, stream>>>(x, ei, s1, E);
    mlp1_gemm_kernel<<<(N + 31) / 32, 256, 0, stream>>>(x, s1, a1, b1f, b2, w2T, h2, N);
    agg2_kernel<<<(E + 7) / 8, 256, 0, stream>>>(h2, ei, s2, E);
    tail_kernel<<<(N + 31) / 32, 256, 0, stream>>>(h2, s2, s3, c3, b4,
                                                   w3T, w4T, wl1T, bl1, wl2, bl2,
                                                   out, N);
}
